// NeRFModel_14963666059930
// MI455X (gfx1250) — matrix operand order repaired and checked by reading, not compiled, not run on hardware
//
#include <hip/hip_runtime.h>
#include <hip/hip_bf16.h>
#include <math.h>

// ---------------------------------------------------------------------------
// NeRF fused forward for MI455X (gfx1250, wave32, WMMA + async-to-LDS).
// One workgroup (8 wave32) per ray: M=128 samples batched through the MLP with
// v_wmma_f32_16x16x32_f16. Weights are pre-packed to f16 B-panel layout by a
// prep kernel; the main kernel streams panels with global_load_async_to_lds
// (ASYNCcnt, double buffered). Panel reads use explicit addrspace(3) pointers
// so B fragments are ds_load_b128 (not flat_load) with immediate offsets.
// ---------------------------------------------------------------------------

typedef __attribute__((ext_vector_type(8)))  _Float16 v8h;
typedef __attribute__((ext_vector_type(16))) _Float16 v16h;
typedef __attribute__((ext_vector_type(8)))  float    v8f;

typedef __attribute__((address_space(3))) _Float16 lds_h;
typedef __attribute__((address_space(3))) v8h      lds_v8h;

#define R_RAYS     1024
#define S_SAMP     128
#define ACT_STRIDE 264   // halves per activation row (256 + pad, multiple of 8)
#define PK         40    // panel K-stride in halves (32 + pad, multiple of 8)

// ---- LDS layout (dynamic shared memory, ~206 KB of the 320 KB WGP pool) ----
constexpr unsigned SM_ACTX   = 0;
constexpr unsigned SM_ACTY   = SM_ACTX   + S_SAMP * ACT_STRIDE * 2;  // 67584
constexpr unsigned SM_ENCP   = SM_ACTY   + S_SAMP * ACT_STRIDE * 2;  // 67584
constexpr unsigned SM_ENCD   = SM_ENCP   + S_SAMP * 64 * 2;          // 16384
constexpr unsigned SM_PANELA = SM_ENCD   + S_SAMP * 32 * 2;          //  8192
constexpr unsigned SM_PANELB = SM_PANELA + 256 * PK * 2;             // 20480
constexpr unsigned SM_PTS    = SM_PANELB + 256 * PK * 2;             // 20480
constexpr unsigned SM_SIG    = SM_PTS    + S_SAMP * 4 * 4;           //  2048
constexpr unsigned SM_DEL    = SM_SIG    + S_SAMP * 4;               //   512
constexpr unsigned SM_COL    = SM_DEL    + S_SAMP * 4;               //   512
constexpr unsigned SMEM_BYTES = SM_COL + S_SAMP * 4 * 4;             // 205824

// ---- packed-weight layout in workspace (halves), per GEMM layer -----------
// layer l: [Kp/32 panels][ncols][32 k-halves], zero-padded past true fan-in.
constexpr size_t PKOFF[10] = {
  0,        // L0: Kp=64,  N=256
  16384,    // L1: Kp=256, N=256
  81920,    // L2
  147456,   // L3
  212992,   // L4: Kp=320, N=256 (skip concat)
  294912,   // L5
  360448,   // L6
  425984,   // L7
  491520,   // feat: Kp=256, N=256
  557056    // color: Kp=288, N=128
};
constexpr size_t PACKED_HALVES = 593920;   // total (1.13 MB as f16)

struct Params {
  const float* inputs;
  const float* Wd[8];
  const float* bd[8];
  const float* w_sigma; const float* b_sigma;
  const float* w_feat;  const float* b_feat;
  const float* Wc;      const float* bc;
  const float* w_rgb;   const float* b_rgb;
  float* out_rgb; float* out_w; float* out_z;
};

// Two aligned 16-byte loads -> one 16-half WMMA fragment (generic, for acts).
__device__ __forceinline__ v16h frag16(const _Float16* p0, const _Float16* p1) {
  v8h lo = *(const v8h*)p0;
  v8h hi = *(const v8h*)p1;
  return __builtin_shufflevector(lo, hi, 0,1,2,3,4,5,6,7,8,9,10,11,12,13,14,15);
}

// Same, but from explicit LDS byte offsets -> guaranteed ds_load_b128.
__device__ __forceinline__ v16h frag16_lds(unsigned off0, unsigned off1) {
  v8h lo = *(const lds_v8h*)off0;
  v8h hi = *(const lds_v8h*)off1;
  return __builtin_shufflevector(lo, hi, 0,1,2,3,4,5,6,7,8,9,10,11,12,13,14,15);
}

template<int N>
__device__ __forceinline__ void wait_async() {
  asm volatile("s_wait_asynccnt %0" :: "n"(N) : "memory");
}

// Issue the async DMA of one packed 32 x (NT*16) f16 panel into LDS.
// Each thread copies (NT*64)/256 16-byte chunks; tracked on ASYNCcnt.
template<int NT>
__device__ __forceinline__ void issue_panel(const _Float16* wpk, int kp, unsigned ldsBase) {
  constexpr int CPT = (NT * 64) / 256;                 // chunks per thread
  const int tid = threadIdx.x;
  unsigned long long gbase =
      (unsigned long long)wpk + (unsigned long long)kp * (NT * 16 * 64);
#pragma unroll
  for (int i = 0; i < CPT; ++i) {
    int c = tid + i * 256;                             // chunk id
    unsigned lds  = ldsBase + (unsigned)(c >> 2) * (PK * 2) + (unsigned)(c & 3) * 16;
    unsigned goff = (unsigned)c * 16;
    asm volatile("global_load_async_to_lds_b128 %0, %1, %2 offset:0"
                 :: "v"(lds), "v"(goff), "s"(gbase)
                 : "memory");
  }
}

// NT WMMAs on one staged panel, with B-fragment rotation (bnext ds_loads issue
// before the WMMA consuming bcur, so LDS reads overlap the matrix pipe).
template<int NT>
__device__ __forceinline__ void compute_panel(v8f* acc, v16h a,
                                              unsigned panOff, int lane) {
  const int laneHi = lane >> 4;
  const int nlo    = lane & 15;
  const unsigned base = panOff + (unsigned)(nlo * PK + laneHi * 16) * 2;
  v16h bcur = frag16_lds(base, base + 16);
#pragma unroll
  for (int nt = 0; nt < NT; ++nt) {
    v16h bnext = bcur;
    if (nt + 1 < NT) {
      unsigned b2 = base + (unsigned)((nt + 1) * 16 * PK * 2);
      bnext = frag16_lds(b2, b2 + 16);
    }
    acc[nt] = __builtin_amdgcn_wmma_f32_16x16x32_f16(
        false, a, false, bcur, (short)0, acc[nt], false, false);
    bcur = bnext;
  }
}

// ---- fallback staging path: convert f32 weights in-kernel -----------------
template<int NT>
__device__ __forceinline__ void gemm_seg_direct(v8f* acc,
    const _Float16* A, int aStride, int segK,
    const float* W, int baseW, int fanin, unsigned panOff)
{
  constexpr int ncols = NT * 16;
  const int tid    = threadIdx.x;
  const int lane   = tid & 31;
  const int wave   = tid >> 5;
  const int laneHi = lane >> 4;
  const int rowA   = wave * 16 + (lane & 15);
  lds_h* pan = (lds_h*)panOff;

  for (int k0 = 0; k0 < segK; k0 += 32) {
    __syncthreads();
    if (tid == 0 && k0 + 32 < segK)
      __builtin_prefetch(W + (size_t)(baseW + k0 + 32) * ncols, 0, 0);
    for (int e = tid; e < 32 * ncols; e += 256) {
      int k = e / ncols, n = e % ncols;
      int wrow = baseW + k0 + k;
      pan[n * PK + k] = (wrow < fanin) ? (_Float16)W[(size_t)wrow * ncols + n]
                                       : (_Float16)0.f;
    }
    __syncthreads();
    const _Float16* ap = A + (size_t)rowA * aStride + k0 + laneHi * 16;
    v16h a = frag16(ap, ap + 8);
    compute_panel<NT>(acc, a, panOff, lane);
  }
}

// ---- async staging path: packed f16 panels via global_load_async_to_lds ---
template<int NT>
__device__ __forceinline__ void gemm_seg_async(v8f* acc,
    const _Float16* A, int aStride, int segK,
    const _Float16* wpk, unsigned ldsA, unsigned ldsB)
{
  constexpr int CPT = (NT * 64) / 256;
  const int lane   = threadIdx.x & 31;
  const int wave   = threadIdx.x >> 5;
  const int laneHi = lane >> 4;
  const int rowA   = wave * 16 + (lane & 15);
  const int np     = segK >> 5;
  const unsigned ldb[2] = { ldsA, ldsB };

  __syncthreads();                       // prior readers of buf0 done
  issue_panel<NT>(wpk, 0, ldb[0]);
  for (int kp = 0; kp < np; ++kp) {
    __syncthreads();                     // prior readers of buf[(kp+1)&1] done
    bool more = (kp + 1 < np);
    if (more) issue_panel<NT>(wpk, kp + 1, ldb[(kp + 1) & 1]);
    if (more) wait_async<CPT>();         // panel kp landed; kp+1 stays in flight
    else      wait_async<0>();
    __syncthreads();                     // panel kp visible to all waves
    const _Float16* ap = A + (size_t)rowA * aStride + kp * 32 + laneHi * 16;
    v16h a = frag16(ap, ap + 8);
    compute_panel<NT>(acc, a, ldb[kp & 1], lane);
  }
}

// Full dense layer: (optional concat second segment) + bias + ReLU + f16 store.
template<int NT, bool RELU, bool ASYNC>
__device__ __attribute__((noinline)) void dense_layer(
    const _Float16* A,  int aStride,  int aK,
    const _Float16* A2, int a2Stride, int a2K,
    const float* W, const _Float16* wpkLayer, int fanin, const float* bias,
    _Float16* out, int outStride, unsigned ldsA, unsigned ldsB)
{
  const int lane   = threadIdx.x & 31;
  const int wave   = threadIdx.x >> 5;
  const int laneHi = lane >> 4;

  v8f acc[NT];
#pragma unroll
  for (int i = 0; i < NT; ++i) acc[i] = v8f{};

  if constexpr (ASYNC) {
    gemm_seg_async<NT>(acc, A, aStride, aK, wpkLayer, ldsA, ldsB);
    if (a2K > 0) {
      const _Float16* wpk2 = wpkLayer + (size_t)(aK >> 5) * (NT * 16 * 32);
      gemm_seg_async<NT>(acc, A2, a2Stride, a2K, wpk2, ldsA, ldsB);
    }
  } else {
    gemm_seg_direct<NT>(acc, A, aStride, aK, W, 0, fanin, ldsA);
    if (a2K > 0)
      gemm_seg_direct<NT>(acc, A2, a2Stride, a2K, W, aK, fanin, ldsA);
  }

#pragma unroll
  for (int nt = 0; nt < NT; ++nt) {
    int col  = nt * 16 + (lane & 15);
    float bc = bias[col];
#pragma unroll
    for (int v = 0; v < 8; ++v) {
      float d = acc[nt][v] + bc;
      if (RELU) d = fmaxf(d, 0.f);
      out[(size_t)(wave * 16 + laneHi * 8 + v) * outStride + col] = (_Float16)d;
    }
  }
  __syncthreads();
}

// ---- prep kernel: pack all GEMM weights to f16 panel layout in workspace --
__global__ void pack_weights(Params p, _Float16* dst) {
  const float* Wsrc[10] = { p.Wd[0], p.Wd[1], p.Wd[2], p.Wd[3], p.Wd[4],
                            p.Wd[5], p.Wd[6], p.Wd[7], p.w_feat, p.Wc };
  const int   Ncol[10] = { 256,256,256,256,256,256,256,256,256,128 };
  const int   Kp  [10] = {  64,256,256,256,320,256,256,256,256,288 };
  const int   Fin [10] = {  60,256,256,256,316,256,256,256,256,280 };

  size_t gtid   = (size_t)blockIdx.x * blockDim.x + threadIdx.x;
  size_t stride = (size_t)gridDim.x * blockDim.x;
  for (int l = 0; l < 10; ++l) {
    const float* W = Wsrc[l];
    const int N = Ncol[l], fi = Fin[l];
    const size_t total = (size_t)Kp[l] * N;
    _Float16* d = dst + PKOFF[l];
    for (size_t idx = gtid; idx < total; idx += stride) {
      int    k    = (int)(idx & 31);
      size_t rest = idx >> 5;
      int    n    = (int)(rest % N);
      int    kp   = (int)(rest / N);
      int    row  = kp * 32 + k;
      d[idx] = (row < fi) ? (_Float16)W[(size_t)row * N + n] : (_Float16)0.f;
    }
  }
}

template<bool ASYNC>
__global__ __launch_bounds__(256, 1) void nerf_fused(Params p, const _Float16* wpk)
{
  extern __shared__ __align__(16) char smem[];
  _Float16* actX  = (_Float16*)(smem + SM_ACTX);
  _Float16* actY  = (_Float16*)(smem + SM_ACTY);
  _Float16* encP  = (_Float16*)(smem + SM_ENCP);    // 128 x 64 (60 used, pad 0)
  _Float16* encD  = (_Float16*)(smem + SM_ENCD);    // 128 x 32 (24 used, pad 0)
  float*    ptsS  = (float*)(smem + SM_PTS);
  float*    sigS  = (float*)(smem + SM_SIG);
  float*    delS  = (float*)(smem + SM_DEL);
  float*    colS  = (float*)(smem + SM_COL);
  // dyn-LDS starts at offset 0 (no static LDS in this TU)
  const unsigned ldsA = SM_PANELA, ldsB = SM_PANELB;

  const int tid = threadIdx.x;
  const int ray = blockIdx.x;
  const float PI = 3.14159265358979323846f;

  // ---- stage 0: load ray, positional encodings into LDS ----
  if (tid < S_SAMP) {
    const float* src = p.inputs + ((size_t)ray * S_SAMP + tid) * 6;
    float px = src[0], py = src[1], pz = src[2];
    ptsS[tid * 4 + 0] = px; ptsS[tid * 4 + 1] = py; ptsS[tid * 4 + 2] = pz;
    float np[3] = { px * 0.25f, py * 0.25f, pz * 0.25f };   // 2*(p+4)/8-1
    float f = PI;
    for (int l = 0; l < 10; ++l) {
      for (int d = 0; d < 3; ++d) {
        float ang = np[d] * f;
        encP[tid * 64 + l * 6 + d]     = (_Float16)sinf(ang);
        encP[tid * 64 + l * 6 + 3 + d] = (_Float16)cosf(ang);
      }
      f *= 2.f;
    }
    for (int j = 60; j < 64; ++j) encP[tid * 64 + j] = (_Float16)0.f;
  } else {
    int s = tid - S_SAMP;
    const float* src = p.inputs + ((size_t)ray * S_SAMP + s) * 6 + 3;
    float nd[3] = { src[0], src[1], src[2] };
    float f = PI;
    for (int l = 0; l < 4; ++l) {
      for (int d = 0; d < 3; ++d) {
        float ang = nd[d] * f;
        encD[s * 32 + l * 6 + d]     = (_Float16)sinf(ang);
        encD[s * 32 + l * 6 + 3 + d] = (_Float16)cosf(ang);
      }
      f *= 2.f;
    }
    for (int j = 24; j < 32; ++j) encD[s * 32 + j] = (_Float16)0.f;
  }
  __syncthreads();

  // ---- deltas (clipped segment lengths) ----
  if (tid < S_SAMP) {
    int j = tid < S_SAMP - 1 ? tid : S_SAMP - 2;
    float dx = ptsS[(j + 1) * 4 + 0] - ptsS[j * 4 + 0];
    float dy = ptsS[(j + 1) * 4 + 1] - ptsS[j * 4 + 1];
    float dz = ptsS[(j + 1) * 4 + 2] - ptsS[j * 4 + 2];
    float d  = sqrtf(dx * dx + dy * dy + dz * dz);
    delS[tid] = fminf(fmaxf(d, 1e-6f), 1e6f);
  }

#define DENSE(NT, RELU, Ain, sA, kA, A2in, sA2, kA2, Wr, pkidx, fi, bi, Oout) \
  dense_layer<NT, RELU, ASYNC>(Ain, sA, kA, A2in, sA2, kA2, Wr,               \
      ASYNC ? (wpk + PKOFF[pkidx]) : nullptr, fi, bi, Oout, ACT_STRIDE,       \
      ldsA, ldsB)

  // ---- density MLP: 8 layers, skip-concat enc_pts into layer 4 ----
  DENSE(16, true , encP, 64, 64,          nullptr, 0, 0,  p.Wd[0], 0,  60, p.bd[0], actX);
  DENSE(16, true , actX, ACT_STRIDE, 256, nullptr, 0, 0,  p.Wd[1], 1, 256, p.bd[1], actY);
  DENSE(16, true , actY, ACT_STRIDE, 256, nullptr, 0, 0,  p.Wd[2], 2, 256, p.bd[2], actX);
  DENSE(16, true , actX, ACT_STRIDE, 256, nullptr, 0, 0,  p.Wd[3], 3, 256, p.bd[3], actY);
  DENSE(16, true , actY, ACT_STRIDE, 256, encP, 64, 64,   p.Wd[4], 4, 316, p.bd[4], actX);
  DENSE(16, true , actX, ACT_STRIDE, 256, nullptr, 0, 0,  p.Wd[5], 5, 256, p.bd[5], actY);
  DENSE(16, true , actY, ACT_STRIDE, 256, nullptr, 0, 0,  p.Wd[6], 6, 256, p.bd[6], actX);
  DENSE(16, true , actX, ACT_STRIDE, 256, nullptr, 0, 0,  p.Wd[7], 7, 256, p.bd[7], actY);

  // ---- sigma head: 256 -> 1 (scalar, reads final activations in actY) ----
  if (tid < S_SAMP) {
    float s = p.b_sigma[0];
    for (int k = 0; k < 256; ++k)
      s += (float)actY[tid * ACT_STRIDE + k] * p.w_sigma[k];
    sigS[tid] = fmaxf(s, 0.f);
  }
  __syncthreads();

  // ---- feature layer (no ReLU) + color layer (concat enc_dirs) ----
  DENSE(16, false, actY, ACT_STRIDE, 256, nullptr, 0, 0, p.w_feat, 8, 256, p.b_feat, actX);
  DENSE( 8, true , actX, ACT_STRIDE, 256, encD, 32, 32,  p.Wc,     9, 280, p.bc,    actY);
#undef DENSE

  // ---- rgb head: 128 -> 3, sigmoid ----
  if (tid < S_SAMP) {
    for (int c = 0; c < 3; ++c) {
      float r = p.b_rgb[c];
      for (int k = 0; k < 128; ++k)
        r += (float)actY[tid * ACT_STRIDE + k] * p.w_rgb[k * 3 + c];
      colS[tid * 4 + c] = 1.f / (1.f + expf(-r));
    }
  }
  __syncthreads();

  // ---- fused volume rendering (serial scan over 128 samples, trivial) ----
  if (tid == 0) {
    float T = 1.f, z = 0.f, rm0 = 0.f, rm1 = 0.f, rm2 = 0.f;
    float* ow = p.out_w + (size_t)ray * S_SAMP;
    float* oz = p.out_z + (size_t)ray * S_SAMP;
    for (int i = 0; i < S_SAMP; ++i) {
      float delta = delS[i];
      float sd    = fminf(sigS[i] * delta, 100.f);
      float alpha = 1.f - expf(-sd);
      float w     = alpha * T;
      ow[i] = w;
      rm0 += w * colS[i * 4 + 0];
      rm1 += w * colS[i * 4 + 1];
      rm2 += w * colS[i * 4 + 2];
      z += delta; oz[i] = z;
      float om = fminf(fmaxf(1.f - alpha, 1e-7f), 1.f);
      T *= om;
    }
    p.out_rgb[ray * 3 + 0] = rm0;
    p.out_rgb[ray * 3 + 1] = rm1;
    p.out_rgb[ray * 3 + 2] = rm2;
  }
}

extern "C" void kernel_launch(void* const* d_in, const int* in_sizes, int n_in,
                              void* d_out, int out_size, void* d_ws, size_t ws_size,
                              hipStream_t stream) {
  (void)in_sizes; (void)n_in; (void)out_size;
  Params p;
  p.inputs = (const float*)d_in[0];
  for (int i = 0; i < 8; ++i) {
    p.Wd[i] = (const float*)d_in[1 + i];
    p.bd[i] = (const float*)d_in[9 + i];
  }
  p.w_sigma = (const float*)d_in[17];
  p.b_sigma = (const float*)d_in[18];
  p.w_feat  = (const float*)d_in[19];
  p.b_feat  = (const float*)d_in[20];
  p.Wc      = (const float*)d_in[21];
  p.bc      = (const float*)d_in[22];
  p.w_rgb   = (const float*)d_in[23];
  p.b_rgb   = (const float*)d_in[24];

  float* out = (float*)d_out;
  p.out_rgb = out;                                // 1024*3
  p.out_w   = out + R_RAYS * 3;                   // 1024*128
  p.out_z   = out + R_RAYS * 3 + R_RAYS * S_SAMP; // 1024*128

  const bool use_async = (d_ws != nullptr) && (ws_size >= PACKED_HALVES * 2);
  if (use_async) {
    _Float16* wpk = (_Float16*)d_ws;
    pack_weights<<<512, 256, 0, stream>>>(p, wpk);
    hipFuncSetAttribute((const void*)nerf_fused<true>,
                        hipFuncAttributeMaxDynamicSharedMemorySize, SMEM_BYTES);
    nerf_fused<true><<<R_RAYS, 256, SMEM_BYTES, stream>>>(p, wpk);
  } else {
    hipFuncSetAttribute((const void*)nerf_fused<false>,
                        hipFuncAttributeMaxDynamicSharedMemorySize, SMEM_BYTES);
    nerf_fused<false><<<R_RAYS, 256, SMEM_BYTES, stream>>>(p, nullptr);
  }
}
